// GraphSAGE_87325275062793
// MI455X (gfx1250) — matrix operand order repaired and checked by reading, not compiled, not run on hardware
//
#include <hip/hip_runtime.h>
#include <hip/hip_fp16.h>

typedef __attribute__((ext_vector_type(16))) _Float16 v16h;
typedef __attribute__((ext_vector_type(8)))  _Float16 v8h;
typedef __attribute__((ext_vector_type(8)))  float    v8f;

#define IN_DIM  128
#define KCAT    256   // concat(mean, x) along K
#define HID_DIM 256
#define OUT_DIM 64

// ---------------------------------------------------------------------------
// Fragment loaders matching CDNA5 WMMA VGPR layouts (CDNA5 ISA §7.12.2)
// A 16x32 f16 (row-major A, ld = row stride in halves):
//   lane<16 : row=lane,    elems 0..7 -> K kb+0..7,  elems 8..15 -> K kb+16..23
//   lane>=16: row=lane-16, elems 0..7 -> K kb+8..15, elems 8..15 -> K kb+24..31
// ---------------------------------------------------------------------------
__device__ __forceinline__ v16h load_fragA(const _Float16* A, int row, int ld,
                                           int kb, int lane) {
    const int khalf = (lane >= 16) ? 8 : 0;
    const _Float16* p = A + (size_t)row * ld + kb + khalf;
    v8h lo = *(const v8h*)(p);
    v8h hi = *(const v8h*)(p + 16);
    return __builtin_shufflevector(lo, hi, 0,1,2,3,4,5,6,7,8,9,10,11,12,13,14,15);
}

// B 32x16 f16, stored K-contiguous (Bt[n][k], ld = K stride):
//   lane<16 : col=lane,    elems 0..7 -> K kb+0..7,  elems 8..15 -> K kb+16..23
//   lane>=16: col=lane-16, elems 0..7 -> K kb+8..15, elems 8..15 -> K kb+24..31
__device__ __forceinline__ v16h load_fragB(const _Float16* Bt, int col, int ld,
                                           int kb, int lane) {
    const int koff = (lane >= 16) ? 8 : 0;
    const _Float16* p = Bt + (size_t)col * ld + kb + koff;
    v8h lo = *(const v8h*)(p);
    v8h hi = *(const v8h*)(p + 16);
    return __builtin_shufflevector(lo, hi, 0,1,2,3,4,5,6,7,8,9,10,11,12,13,14,15);
}

// ---------------------------------------------------------------------------
// Phase 0: zero agg[N*128] and cnt[N]
// ---------------------------------------------------------------------------
__global__ void sage_zero(float* __restrict__ agg, float* __restrict__ cnt,
                          int n_agg, int n_cnt) {
    int i = blockIdx.x * blockDim.x + threadIdx.x;
    if (i < n_agg)           agg[i] = 0.0f;
    else if (i < n_agg + n_cnt) cnt[i - n_agg] = 0.0f;
}

// ---------------------------------------------------------------------------
// Phase 1: edge scatter. One wave32 per edge; lane l handles floats 4l..4l+3.
// x rows (25.6MB) and agg (25.6MB) both live in the 192MB L2 -> L2-rate atomics.
// ---------------------------------------------------------------------------
__global__ void sage_scatter(const float* __restrict__ x,
                             const long long* __restrict__ ei,
                             float* __restrict__ agg, float* __restrict__ cnt,
                             int E) {
    int gid  = blockIdx.x * blockDim.x + threadIdx.x;
    int e    = gid >> 5;
    int lane = gid & 31;
    if (e >= E) return;
    long long src = ei[e];                 // edge_index[0][e]
    long long dst = ei[(long long)E + e];  // edge_index[1][e]
    const float4 v = ((const float4*)(x + src * IN_DIM))[lane];
    float* a = agg + dst * IN_DIM + lane * 4;
    __hip_atomic_fetch_add(a + 0, v.x, __ATOMIC_RELAXED, __HIP_MEMORY_SCOPE_AGENT);
    __hip_atomic_fetch_add(a + 1, v.y, __ATOMIC_RELAXED, __HIP_MEMORY_SCOPE_AGENT);
    __hip_atomic_fetch_add(a + 2, v.z, __ATOMIC_RELAXED, __HIP_MEMORY_SCOPE_AGENT);
    __hip_atomic_fetch_add(a + 3, v.w, __ATOMIC_RELAXED, __HIP_MEMORY_SCOPE_AGENT);
    if (lane == 0)
        __hip_atomic_fetch_add(cnt + dst, 1.0f, __ATOMIC_RELAXED, __HIP_MEMORY_SCOPE_AGENT);
}

// ---------------------------------------------------------------------------
// Phase 2a: A1[i][k] = f16( k<128 ? agg[i][k]/max(cnt[i],1) : x[i][k-128] )
// ---------------------------------------------------------------------------
__global__ void sage_build_a(const float* __restrict__ x,
                             const float* __restrict__ agg,
                             const float* __restrict__ cnt,
                             _Float16* __restrict__ A1, int N) {
    int idx = blockIdx.x * blockDim.x + threadIdx.x;
    if (idx >= N * KCAT) return;
    int i = idx / KCAT, k = idx % KCAT;
    float v;
    if (k < IN_DIM) {
        float c = cnt[i];
        v = agg[(size_t)i * IN_DIM + k] / fmaxf(c, 1.0f);
    } else {
        v = x[(size_t)i * IN_DIM + (k - IN_DIM)];
    }
    A1[idx] = (_Float16)v;
}

// Phase 2b: transpose+convert weights to K-contiguous f16.
// Bt[n][k] = k<128 ? W_l[k][n] : W_r[k-128][n]   (256x256)
// WoT[n][k] = W_out[k][n]                         (64x256)
__global__ void sage_build_w(const float* __restrict__ W_l,
                             const float* __restrict__ W_r,
                             const float* __restrict__ W_out,
                             _Float16* __restrict__ Bt,
                             _Float16* __restrict__ WoT) {
    int t = blockIdx.x * blockDim.x + threadIdx.x;
    if (t < HID_DIM * KCAT) {
        int n = t / KCAT, k = t % KCAT;
        float v = (k < IN_DIM) ? W_l[(size_t)k * HID_DIM + n]
                               : W_r[(size_t)(k - IN_DIM) * HID_DIM + n];
        Bt[t] = (_Float16)v;
    } else if (t < HID_DIM * KCAT + OUT_DIM * HID_DIM) {
        int t2 = t - HID_DIM * KCAT;
        int n = t2 / HID_DIM, k = t2 % HID_DIM;
        WoT[t2] = (_Float16)W_out[(size_t)k * OUT_DIM + n];
    }
}

// ---------------------------------------------------------------------------
// Phase 3: H = elu(A1[N,256] @ Bt^T + b_l), f16 out.  One 16x16 tile per wave.
// 256 threads = 8 waves per block; tile guard is wave-uniform (EXEC stays ~0/~full).
// ---------------------------------------------------------------------------
__global__ void sage_gemm1(const _Float16* __restrict__ A1,
                           const _Float16* __restrict__ Bt,
                           const float* __restrict__ b_l,
                           _Float16* __restrict__ H, int tilesTotal) {
    int wave = (blockIdx.x * blockDim.x + threadIdx.x) >> 5;
    int lane = threadIdx.x & 31;
    if (wave >= tilesTotal) return;            // uniform per wave
    int tm = wave >> 4;                        // 16 N-tiles (HID 256/16)
    int tn = wave & 15;
    int row = tm * 16 + (lane & 15);
    int col = tn * 16 + (lane & 15);
    v8f acc = {};
#pragma unroll
    for (int kb = 0; kb < KCAT; kb += 32) {
        v16h a = load_fragA(A1, row, KCAT, kb, lane);
        v16h b = load_fragB(Bt, col, KCAT, kb, lane);
        acc = __builtin_amdgcn_wmma_f32_16x16x32_f16(false, a, false, b,
                                                     (short)0, acc, false, false);
    }
    float bias = b_l[col];
    int rbase = tm * 16 + ((lane >= 16) ? 8 : 0); // C/D: VGPR r -> row r + 8*(lane>=16)
#pragma unroll
    for (int r = 0; r < 8; ++r) {
        float v = acc[r] + bias;
        v = (v > 0.0f) ? v : (__expf(v) - 1.0f);  // ELU
        H[(size_t)(rbase + r) * HID_DIM + col] = (_Float16)v;
    }
}

// ---------------------------------------------------------------------------
// Phase 4: out = H[N,256] @ WoT^T + b_out, f32 out. 128 threads = 4 waves/block.
// ---------------------------------------------------------------------------
__global__ void sage_gemm2(const _Float16* __restrict__ H,
                           const _Float16* __restrict__ WoT,
                           const float* __restrict__ b_out,
                           float* __restrict__ out, int tilesTotal) {
    int wave = (blockIdx.x * blockDim.x + threadIdx.x) >> 5;
    int lane = threadIdx.x & 31;
    if (wave >= tilesTotal) return;            // uniform per wave
    int tm = wave >> 2;                        // 4 N-tiles (OUT 64/16)
    int tn = wave & 3;
    int row = tm * 16 + (lane & 15);
    int col = tn * 16 + (lane & 15);
    v8f acc = {};
#pragma unroll
    for (int kb = 0; kb < HID_DIM; kb += 32) {
        v16h a = load_fragA(H, row, HID_DIM, kb, lane);
        v16h b = load_fragB(WoT, col, HID_DIM, kb, lane);
        acc = __builtin_amdgcn_wmma_f32_16x16x32_f16(false, a, false, b,
                                                     (short)0, acc, false, false);
    }
    float bias = b_out[col];
    int rbase = tm * 16 + ((lane >= 16) ? 8 : 0);
#pragma unroll
    for (int r = 0; r < 8; ++r)
        out[(size_t)(rbase + r) * OUT_DIM + col] = acc[r] + bias;
}

// ---------------------------------------------------------------------------
extern "C" void kernel_launch(void* const* d_in, const int* in_sizes, int n_in,
                              void* d_out, int out_size, void* d_ws, size_t ws_size,
                              hipStream_t stream) {
    const float*     x     = (const float*)d_in[0];
    const long long* ei    = (const long long*)d_in[1];   // int64 edge_index [2,E]
    const float*     W_l   = (const float*)d_in[2];
    const float*     b_l   = (const float*)d_in[3];
    const float*     W_r   = (const float*)d_in[4];
    const float*     W_out = (const float*)d_in[5];
    const float*     b_out = (const float*)d_in[6];
    float*           out   = (float*)d_out;

    const int N = in_sizes[0] / IN_DIM;   // 50000 (divisible by 16)
    const int E = in_sizes[1] / 2;        // 800000

    // Workspace carve-up (256B aligned): agg f32, cnt f32, A1 f16, H f16, Bt f16, WoT f16
    char* ws = (char*)d_ws;
    size_t off = 0;
    auto carve = [&](size_t bytes) -> void* {
        void* p = ws + off;
        off += (bytes + 255) & ~(size_t)255;
        return p;
    };
    float*     agg = (float*)    carve((size_t)N * IN_DIM * sizeof(float));
    float*     cnt = (float*)    carve((size_t)N * sizeof(float));
    _Float16*  A1  = (_Float16*) carve((size_t)N * KCAT * sizeof(_Float16));
    _Float16*  H   = (_Float16*) carve((size_t)N * HID_DIM * sizeof(_Float16));
    _Float16*  Bt  = (_Float16*) carve((size_t)HID_DIM * KCAT * sizeof(_Float16));
    _Float16*  WoT = (_Float16*) carve((size_t)OUT_DIM * HID_DIM * sizeof(_Float16));

    const int TB = 256;

    // Phase 0: zero accumulators
    {
        int total = N * IN_DIM + N;
        sage_zero<<<(total + TB - 1) / TB, TB, 0, stream>>>(agg, cnt, N * IN_DIM, N);
    }
    // Phase 2b can run concurrently-in-order; launch early (independent of scatter)
    {
        int total = HID_DIM * KCAT + OUT_DIM * HID_DIM;
        sage_build_w<<<(total + TB - 1) / TB, TB, 0, stream>>>(W_l, W_r, W_out, Bt, WoT);
    }
    // Phase 1: edge scatter (one wave per edge)
    {
        long long threads = (long long)E * 32;
        sage_scatter<<<(int)((threads + TB - 1) / TB), TB, 0, stream>>>(x, ei, agg, cnt, E);
    }
    // Phase 2a: build concat activation matrix in f16
    {
        int total = N * KCAT;
        sage_build_a<<<(total + TB - 1) / TB, TB, 0, stream>>>(x, agg, cnt, A1, N);
    }
    // Phase 3: fused GEMM1 + bias + ELU
    {
        int tiles = (N / 16) * (HID_DIM / 16);             // 3125 * 16
        int blocks = (tiles * 32 + TB - 1) / TB;           // 8 waves/block
        sage_gemm1<<<blocks, TB, 0, stream>>>(A1, Bt, b_l, H, tiles);
    }
    // Phase 4: GEMM2 + bias
    {
        int tiles = (N / 16) * (OUT_DIM / 16);             // 3125 * 4
        int blocks = (tiles * 32 + 127) / 128;             // 4 waves/block
        sage_gemm2<<<blocks, 128, 0, stream>>>(H, WoT, b_out, out, tiles);
    }
}
